// LlamaAttention_76287209111756
// MI455X (gfx1250) — matrix operand order repaired and checked by reading, not compiled, not run on hardware
//
#include <hip/hip_runtime.h>
#include <hip/hip_bf16.h>
#include <stdint.h>

// ---------------------------------------------------------------------------
// LLaMA attention forward for MI455X (gfx1250, wave32, WMMA).
// Outputs: d_out = [ out (2048*2048 f32) | attn (32*2048*2048 f32) ]
// ---------------------------------------------------------------------------

#define DIM      2048
#define T_SEQ    2048
#define N_HEAD   32
#define N_KVH    8
#define HEAD_DIM 64
#define KV_DIM   (N_KVH * HEAD_DIM)   // 512

typedef __attribute__((ext_vector_type(16))) _Float16 v16h;
typedef __attribute__((ext_vector_type(8)))  _Float16 v8h;
typedef __attribute__((ext_vector_type(8)))  float    v8f;

union Frag16 { v16h v; v8h h[2]; };

__device__ __forceinline__ v8f wmma_f16(v16h a, v16h b, v8f c) {
  // D(f32 16x16) = A(f16 16x32) * B(f16 32x16) + C   -> v_wmma_f32_16x16x32_f16
  return __builtin_amdgcn_wmma_f32_16x16x32_f16(
      /*neg_a=*/false, a, /*neg_b=*/false, b,
      /*c_mod=*/(short)0, c, /*reuse_a=*/false, /*reuse_b=*/false);
}

// ---------------------------------------------------------------------------
// f32 -> f16 elementwise conversion
// ---------------------------------------------------------------------------
__global__ void cvt_f32_to_f16(const float* __restrict__ in,
                               _Float16* __restrict__ out, int n) {
  int i = blockIdx.x * blockDim.x + threadIdx.x;
  if (i < n) out[i] = (_Float16)in[i];
}

// ---------------------------------------------------------------------------
// Tiled WMMA GEMM: C[M,N] = A[M,K] * B[K,N]; A,B f16 row-major.
// 256 threads = 8 waves; 64x64 tile; each wave computes a 16x32 strip
// (two 16x16 WMMA accumulators). k-step 32 staged through LDS.
// storeHalf: 0 -> Cf (f32), 1 -> Ch (f16)
// ---------------------------------------------------------------------------
__global__ __launch_bounds__(256) void gemm_wmma(
    const _Float16* __restrict__ A, const _Float16* __restrict__ B,
    float* __restrict__ Cf, _Float16* __restrict__ Ch,
    int M, int N, int K, int storeHalf)
{
  __shared__ _Float16 As[64][32];    // [m][k]
  __shared__ _Float16 BsT[64][32];   // [n][k] (transposed on the way in)

  const int tid  = threadIdx.x;
  const int lane = tid & 31;
  const int wave = tid >> 5;         // 0..7
  const int wr   = wave >> 1;        // 0..3 : 16-row group
  const int wc   = wave & 1;         // 0..1 : 32-col group
  const int half = lane >> 4;        // A/B half-lane split (ISA 7.12.2)
  const int l15  = lane & 15;

  const int m0 = blockIdx.y * 64;
  const int n0 = blockIdx.x * 64;

  v8f acc0 = {}; v8f acc1 = {};

  for (int k0 = 0; k0 < K; k0 += 32) {
    __syncthreads();
    { // A tile: 64x32, 8 contiguous f16 per thread
      int r = tid >> 2, c = (tid & 3) * 8;
      *(v8h*)&As[r][c] = *(const v8h*)(A + (size_t)(m0 + r) * K + k0 + c);
    }
    { // B tile: 32x64 read row-major, stored transposed [n][k]
      int r = tid >> 3, c = (tid & 7) * 8;
      v8h t = *(const v8h*)(B + (size_t)(k0 + r) * N + n0 + c);
      #pragma unroll
      for (int j = 0; j < 8; ++j) BsT[c + j][r] = t[j];
    }
    if (k0 + 32 < K) { // prefetch next A tile cacheline path (global_prefetch_b8)
      __builtin_prefetch(A + (size_t)(m0 + (tid >> 2)) * K + k0 + 32, 0, 1);
    }
    __syncthreads();

    // A fragment: row m, K-halves {8h..8h+7} and {16+8h..16+8h+7}
    Frag16 a;
    int mrow = l15 + wr * 16;
    a.h[0] = *(const v8h*)&As[mrow][8 * half];
    a.h[1] = *(const v8h*)&As[mrow][16 + 8 * half];

    // B fragments: column n = l15 (+16 for second tile), K-halves {8h..8h+7} and {16+8h..16+8h+7}
    int kk = half * 8;
    Frag16 b0, b1;
    int nc0 = l15 + wc * 32;
    b0.h[0] = *(const v8h*)&BsT[nc0][kk];
    b0.h[1] = *(const v8h*)&BsT[nc0][kk + 16];
    b1.h[0] = *(const v8h*)&BsT[nc0 + 16][kk];
    b1.h[1] = *(const v8h*)&BsT[nc0 + 16][kk + 16];

    acc0 = wmma_f16(a.v, b0.v, acc0);
    acc1 = wmma_f16(a.v, b1.v, acc1);
  }

  // C layout: VGPR i -> row (i + 8*half), col = l15
  const int rbase = m0 + wr * 16 + half * 8;
  const int c0 = n0 + wc * 32 + l15;
  #pragma unroll
  for (int i = 0; i < 8; ++i) {
    int r = rbase + i;
    if (storeHalf) {
      Ch[(size_t)r * N + c0]      = (_Float16)acc0[i];
      Ch[(size_t)r * N + c0 + 16] = (_Float16)acc1[i];
    } else {
      Cf[(size_t)r * N + c0]      = acc0[i];
      Cf[(size_t)r * N + c0 + 16] = acc1[i];
    }
  }
}

// ---------------------------------------------------------------------------
// RoPE (interleaved pairs) + f32->f16. Layout [T][C], rotate each 64-block.
// grid = (T, C/64), block = 32 (one lane per pair).
// ---------------------------------------------------------------------------
__global__ void rope_f32_to_f16(const float* __restrict__ in,
                                _Float16* __restrict__ out, int C) {
  const int t = blockIdx.x;
  const int blk = blockIdx.y;
  const int i = threadIdx.x;  // pair index 0..31
  const float* src = in + (size_t)t * C + blk * 64;
  _Float16*   dst  = out + (size_t)t * C + blk * 64;
  float x1 = src[2 * i], x2 = src[2 * i + 1];
  float theta = __expf(-((float)(2 * i) / 64.0f) * __logf(10000.0f));
  float ang = (float)t * theta;
  float c = __cosf(ang), s = __sinf(ang);
  dst[2 * i]     = (_Float16)(x1 * c - x2 * s);
  dst[2 * i + 1] = (_Float16)(x1 * s + x2 * c);
}

// ---------------------------------------------------------------------------
// Fused attention: one WG per (head, 16 query rows). 128 threads = 4 waves.
// Phase 1: S = Q K^T via WMMA into 16x2048 f32 LDS (uses CDNA5's 320KB LDS).
// Phase 2: softmax in LDS (mask quirk applied in Phase 1), stream attn f32 to global once.
// Phase 3: P·V via WMMA, P converted f32->f16 on the fly from LDS.
// ---------------------------------------------------------------------------
#define SPAD (T_SEQ + 4)   // row pad to break 64-bank alignment

__global__ __launch_bounds__(128) void attn_fused(
    const _Float16* __restrict__ Qh,   // [T][2048]  (RoPE'd)
    const _Float16* __restrict__ Kh,   // [T][512]   (RoPE'd)
    const _Float16* __restrict__ Vh,   // [T][512]
    const int*      __restrict__ mask, // [T][T]
    float*          __restrict__ attn, // [32][T][T]
    _Float16*       __restrict__ ctx)  // [T][2048]
{
  __shared__ float S[16][SPAD];        // ~128.25 KB
  __shared__ float red[16][8];
  __shared__ float rowmax[16], rowsum[16];

  const int h  = blockIdx.y;
  const int qb = blockIdx.x * 16;
  const int hk = h >> 2;               // GQA: 4 Q heads per KV head
  const int tid = threadIdx.x;
  const int lane = tid & 31, wave = tid >> 5;   // 4 waves
  const int half = lane >> 4, l15 = lane & 15;

  // --- Q fragments (16 rows x 64 d, split into two K=32 A-fragments) ---
  Frag16 aq0, aq1;
  {
    const _Float16* qr = Qh + (size_t)(qb + l15) * DIM + h * HEAD_DIM;
    aq0.h[0] = *(const v8h*)(qr + 8 * half);
    aq0.h[1] = *(const v8h*)(qr + 16 + 8 * half);
    aq1.h[0] = *(const v8h*)(qr + 32 + 8 * half);
    aq1.h[1] = *(const v8h*)(qr + 48 + 8 * half);
  }

  const float scale = 0.125f;          // 1/sqrt(64)

  // --- Phase 1: scores, 128 key-tiles of 16, round-robin over waves ---
  for (int kt = wave; kt < T_SEQ / 16; kt += 4) {
    int kc = kt * 16 + l15;            // this lane's key column
    const _Float16* kr = Kh + (size_t)kc * KV_DIM + hk * HEAD_DIM;
    int kd = half * 8;
    Frag16 b0, b1;                     // B = K^T : col = key, k-dim = d
    b0.h[0] = *(const v8h*)(kr + kd);
    b0.h[1] = *(const v8h*)(kr + kd + 16);
    b1.h[0] = *(const v8h*)(kr + 32 + kd);
    b1.h[1] = *(const v8h*)(kr + 32 + kd + 16);
    v8f acc = {};
    acc = wmma_f16(aq0.v, b0.v, acc);
    acc = wmma_f16(aq1.v, b1.v, acc);
    #pragma unroll
    for (int i = 0; i < 8; ++i) {
      int qr_l = i + 8 * half;
      int q = qb + qr_l;
      int k = kt * 16 + l15;
      float s = acc[i] * scale;
      // faithful: scores += where(mask==0, -1e9, scores)
      s = (mask[(size_t)q * T_SEQ + k] == 0) ? (s - 1e9f) : (2.0f * s);
      S[qr_l][k] = s;
    }
  }
  __syncthreads();

  // --- Phase 2: softmax over each of 16 rows; 8 threads per row ---
  {
    int r = tid >> 3, sub = tid & 7;
    int kb = sub * (T_SEQ / 8), ke = kb + (T_SEQ / 8);
    float m = -3.4e38f;
    for (int k = kb; k < ke; ++k) m = fmaxf(m, S[r][k]);
    red[r][sub] = m;
    __syncthreads();
    if (sub == 0) {
      float mm = red[r][0];
      #pragma unroll
      for (int j = 1; j < 8; ++j) mm = fmaxf(mm, red[r][j]);
      rowmax[r] = mm;
    }
    __syncthreads();
    float mm = rowmax[r], sum = 0.f;
    for (int k = kb; k < ke; ++k) {
      float e = __expf(S[r][k] - mm);
      S[r][k] = e;
      sum += e;
    }
    red[r][sub] = sum;
    __syncthreads();
    if (sub == 0) {
      float ss = 0.f;
      #pragma unroll
      for (int j = 0; j < 8; ++j) ss += red[r][j];
      rowsum[r] = ss;
    }
    __syncthreads();
    float inv = 1.0f / rowsum[r];
    float* arow = attn + ((size_t)h * T_SEQ + (qb + r)) * T_SEQ;
    for (int k = kb; k < ke; ++k) {
      float p = S[r][k] * inv;
      S[r][k] = p;
      arow[k] = p;                     // single streamed attn write
    }
  }
  __syncthreads();

  // --- Phase 3: ctx = P @ V ; wave owns d-tile [wave*16, wave*16+16) ---
  {
    v8f acc = {};
    const int d = hk * HEAD_DIM + wave * 16 + l15;   // column in Vh row
    for (int kk = 0; kk < T_SEQ / 32; ++kk) {
      Frag16 pa;                       // A = P (16x32), f32->f16 from LDS
      #pragma unroll
      for (int j = 0; j < 8; ++j) {
        pa.v[j]     = (_Float16)S[l15][kk * 32 + 8 * half + j];
        pa.v[8 + j] = (_Float16)S[l15][kk * 32 + 16 + 8 * half + j];
      }
      Frag16 vb;                       // B = V (32x16): col d, k-rows {8h..8h+7} and {16+8h..16+8h+7}
      int kr0 = kk * 32 + half * 8;
      #pragma unroll
      for (int r2 = 0; r2 < 16; ++r2)
        vb.v[r2] = Vh[(size_t)(kr0 + (r2 & 7) + (r2 >> 3) * 16) * KV_DIM + d];
      acc = wmma_f16(pa.v, vb.v, acc);
    }
    #pragma unroll
    for (int i = 0; i < 8; ++i) {
      int q = qb + i + 8 * half;
      ctx[(size_t)q * DIM + h * HEAD_DIM + wave * 16 + l15] = (_Float16)acc[i];
    }
  }
}

// ---------------------------------------------------------------------------
// Host-side launch
// ---------------------------------------------------------------------------
extern "C" void kernel_launch(void* const* d_in, const int* in_sizes, int n_in,
                              void* d_out, int out_size, void* d_ws, size_t ws_size,
                              hipStream_t stream) {
  const float* x    = (const float*)d_in[0];
  const float* Wq   = (const float*)d_in[1];
  const float* Wk   = (const float*)d_in[2];
  const float* Wv   = (const float*)d_in[3];
  const float* Wo   = (const float*)d_in[4];
  const int*   mask = (const int*)d_in[5];

  float* out_p  = (float*)d_out;                              // [T][DIM]
  float* attn_p = (float*)d_out + (size_t)T_SEQ * DIM;        // [H][T][T]

  char* ws = (char*)d_ws;
  const size_t MB = 1024 * 1024;
  _Float16* xb   = (_Float16*)(ws + 0 * MB);    // 8 MB  [T][DIM]
  _Float16* Wqb  = (_Float16*)(ws + 8 * MB);    // 8 MB
  _Float16* Wkb  = (_Float16*)(ws + 16 * MB);   // 2 MB
  _Float16* Wvb  = (_Float16*)(ws + 18 * MB);   // 2 MB
  _Float16* Wob  = (_Float16*)(ws + 20 * MB);   // 8 MB
  float*    Qf   = (float*)   (ws + 28 * MB);   // 16 MB [T][DIM]
  float*    Kf   = (float*)   (ws + 44 * MB);   // 4 MB  [T][512]
  _Float16* Qh   = (_Float16*)(ws + 48 * MB);   // 8 MB
  _Float16* Kh   = (_Float16*)(ws + 56 * MB);   // 2 MB
  _Float16* Vh   = (_Float16*)(ws + 58 * MB);   // 2 MB
  _Float16* ctx  = (_Float16*)(ws + 60 * MB);   // 8 MB

  // 1) f32 -> f16 conversions
  auto cvt = [&](const float* src, _Float16* dst, int n) {
    cvt_f32_to_f16<<<(n + 255) / 256, 256, 0, stream>>>(src, dst, n);
  };
  cvt(x,  xb,  T_SEQ * DIM);
  cvt(Wq, Wqb, DIM * DIM);
  cvt(Wk, Wkb, DIM * KV_DIM);
  cvt(Wv, Wvb, DIM * KV_DIM);
  cvt(Wo, Wob, DIM * DIM);

  // 2) Projections (WMMA GEMMs)
  gemm_wmma<<<dim3(DIM / 64,    T_SEQ / 64), 256, 0, stream>>>(
      xb, Wqb, Qf, nullptr, T_SEQ, DIM, DIM, 0);
  gemm_wmma<<<dim3(KV_DIM / 64, T_SEQ / 64), 256, 0, stream>>>(
      xb, Wkb, Kf, nullptr, T_SEQ, KV_DIM, DIM, 0);
  gemm_wmma<<<dim3(KV_DIM / 64, T_SEQ / 64), 256, 0, stream>>>(
      xb, Wvb, nullptr, Vh, T_SEQ, KV_DIM, DIM, 1);

  // 3) RoPE + f16
  rope_f32_to_f16<<<dim3(T_SEQ, DIM / 64),    32, 0, stream>>>(Qf, Qh, DIM);
  rope_f32_to_f16<<<dim3(T_SEQ, KV_DIM / 64), 32, 0, stream>>>(Kf, Kh, KV_DIM);

  // 4) Fused attention (scores + mask + softmax + attn write + P·V)
  attn_fused<<<dim3(T_SEQ / 16, N_HEAD), 128, 0, stream>>>(
      Qh, Kh, Vh, mask, attn_p, ctx);

  // 5) Output projection
  gemm_wmma<<<dim3(DIM / 64, T_SEQ / 64), 256, 0, stream>>>(
      ctx, Wob, out_p, nullptr, T_SEQ, DIM, DIM, 0);
}
